// Decompressor_33432025432125
// MI455X (gfx1250) — matrix operand order repaired and checked by reading, not compiled, not run on hardware
//
#include <hip/hip_runtime.h>

// ---------------------------------------------------------------------------
// Autoregressive LSTM (B=64, H=1024, T=256) + LayerNorm + projection to D=512
// Strategy:
//   * W_comb = W_ih + W_hh (f16) since input == h for t>=1.
//   * Persistent recurrence kernel: 64 blocks x 512 threads (16 waves).
//     Block owns 16 columns of H -> 16 WMMA 16x16 tiles (4 gates x 4 M-tiles).
//     Weight slice (128KB) resident in LDS; h staged to LDS each step (128KB)
//     via CDNA5 global_load_async_to_lds_b128 (ASYNCcnt path, no VGPR round
//     trip); cell state + gate exchange in LDS. Grid barrier via global atomic.
//   * LayerNorm kernel, then WMMA GEMM for the output projection.
// ---------------------------------------------------------------------------

typedef __attribute__((ext_vector_type(16))) _Float16 v16h;
typedef __attribute__((ext_vector_type(8)))  _Float16 v8h;
typedef __attribute__((ext_vector_type(8)))  float    v8f;

#define BATCH   64
#define HDIM    1024
#define GDIM    4096   // 4*H
#define TSTEPS  256
#define DDIM    512
#define NBLK    64     // blocks in recurrence kernel (each owns 16 cols of H)
#define NTHR    512    // 16 waves

// LDS layout for the persistent kernel (bytes)
#define SM_HS   0                                  // h stage: 64*1024 f16
#define SM_WS   (64 * 1024 * 2)                    // W slice: 64*1024 f16
#define SM_GS   (SM_WS + 64 * 1024 * 2)            // gates:   4*64*16 f32
#define SM_CS   (SM_GS + 4 * 64 * 16 * 4)          // c state: 64*16 f32
#define SM_TOT  (SM_CS + 64 * 16 * 4)              // = 282624 bytes

// Workspace layout (bytes, all 256-aligned). Total ~76.9 MB.
#define OFF_WC    ((size_t)0)                       //  4096*1024 f16 = 8 MB
#define OFF_SEQ   (OFF_WC   + (size_t)GDIM*HDIM*2)  // 16384*1024 f16 = 32 MB
#define OFF_SEQN  (OFF_SEQ  + (size_t)BATCH*TSTEPS*HDIM*2)
#define OFF_WO    (OFF_SEQN + (size_t)BATCH*TSTEPS*HDIM*2) // 512*1024 f16
#define OFF_HG0   (OFF_WO   + (size_t)DDIM*HDIM*2)
#define OFF_HG1   (OFF_HG0  + (size_t)BATCH*HDIM*2)
#define OFF_SYNC  (OFF_HG1  + (size_t)BATCH*HDIM*2)
#define OFF_BCOMB (OFF_SYNC + 256)                  // 4096 f32

// ----- helpers --------------------------------------------------------------

// A fragment (16x32 f16, MxK): caller passes &row[k0 + kA]; elems 0..15 are
// contiguous k from base (k = 16h + s, matching the B operand).
__device__ inline v16h load_a16(const _Float16* base) {
  v8h lo = *(const v8h*)(base);
  v8h hi = *(const v8h*)(base + 8);
  v16h a;
#pragma unroll
  for (int i = 0; i < 8; ++i) { a[i] = lo[i]; a[i + 8] = hi[i]; }
  return a;
}

__device__ inline float sigmoidf_(float x) { return 1.f / (1.f + __expf(-x)); }

// Low 32 bits of a generic pointer into LDS = LDS byte offset (aperture puts
// the shared-base tag in the high 32 bits).
__device__ inline unsigned lds_off(const void* p) {
  return (unsigned)(uintptr_t)p;
}

// CDNA5 async copy: 16B global -> LDS, tracked by ASYNCcnt (GV addressing:
// LDS dest offset VGPR, 64-bit global address VGPR pair).
__device__ inline void async_cp_b128(unsigned lds, const void* gptr) {
  asm volatile("global_load_async_to_lds_b128 %0, %1, off"
               :: "v"(lds), "v"(gptr) : "memory");
}

__device__ inline void wait_async0() {
  asm volatile("s_wait_asynccnt 0x0" ::: "memory");
}

// ----- prep kernels ---------------------------------------------------------

__global__ void prep_wcomb(const float* __restrict__ Wih,
                           const float* __restrict__ Whh,
                           _Float16* __restrict__ Wc, int n) {
  int i = blockIdx.x * 256 + threadIdx.x;
  if (i < n) Wc[i] = (_Float16)(Wih[i] + Whh[i]);
}

__global__ void prep_misc(const float* __restrict__ x,
                          const float* __restrict__ bih,
                          const float* __restrict__ bhh,
                          const float* __restrict__ Wout,
                          _Float16* __restrict__ hG0,
                          float* __restrict__ bcomb,
                          _Float16* __restrict__ Wo16,
                          unsigned* __restrict__ sync) {
  int i = blockIdx.x * 256 + threadIdx.x;
  if (i < DDIM * HDIM)   Wo16[i] = (_Float16)Wout[i];
  if (i < BATCH * HDIM)  hG0[i]  = (_Float16)x[i];
  if (i < GDIM)          bcomb[i] = bih[i] + bhh[i];
  if (i == 0)            *sync = 0u;
}

// ----- persistent LSTM recurrence -------------------------------------------

__global__ __launch_bounds__(NTHR, 1)
void lstm_persistent(const float*    __restrict__ W_ih,   // f32 [4096][1024] (step 0 B)
                     const _Float16* __restrict__ Wc,     // f16 [4096][1024]
                     const float*    __restrict__ bcomb,  // [4096]
                     _Float16*       __restrict__ hG0,
                     _Float16*       __restrict__ hG1,
                     _Float16*       __restrict__ seq16,  // [64][256][1024]
                     unsigned*       __restrict__ sync) {
  extern __shared__ char smem[];
  _Float16* hS = (_Float16*)(smem + SM_HS);   // [64][1024]
  _Float16* wS = (_Float16*)(smem + SM_WS);   // [64 local outcols][1024]
  float*    gS = (float*)(smem + SM_GS);      // [4][64][16]
  float*    cS = (float*)(smem + SM_CS);      // [64][16]

  const int tid  = threadIdx.x;
  const int n0   = blockIdx.x * 16;           // this block's H-column base
  const int lane = tid & 31;
  const int wave = tid >> 5;
  const int mt   = wave & 3;                  // M tile (batch rows 16*mt..)
  const int gate = wave >> 2;                 // 0=i 1=f 2=g 3=o

  const unsigned hS_base = lds_off(hS);

  // Async-load constant weight slice into LDS: local row lc -> global row
  // (lc>>4)*1024 + n0 + (lc&15). 8 threads per row, 128 f16 (16 x B128) each.
  {
    int lc  = tid >> 3;
    int seg = (tid & 7) * 128;
    const char* src =
        (const char*)(Wc + (size_t)((lc >> 4) * HDIM + n0 + (lc & 15)) * HDIM + seg);
    unsigned dst = lds_off(wS + lc * HDIM + seg);
#pragma unroll
    for (int j = 0; j < 16; ++j)
      async_cp_b128(dst + (unsigned)(j * 16), src + (size_t)j * 16);
  }
  for (int p = tid; p < BATCH * 16; p += NTHR) cS[p] = 0.f;

  for (int t = 0; t < TSTEPS; ++t) {
    const _Float16* hin  = (t & 1) ? hG1 : hG0;
    _Float16*       hout = (t & 1) ? hG0 : hG1;

    // Stage full h (64x1024 f16 = 8192 x B128) into LDS via async copy.
    {
      const char* src = (const char*)hin + tid * 16;
      unsigned dst = hS_base + (unsigned)(tid * 16);
#pragma unroll
      for (int j = 0; j < 16; ++j)
        async_cp_b128(dst + (unsigned)(j * NTHR * 16), src + (size_t)j * NTHR * 16);
    }
    wait_async0();      // __syncthreads only covers DScnt, not ASYNCcnt
    __syncthreads();

    // 16x16 tile of gates via WMMA, K = 1024.
    v8f acc = {};
    const int m  = mt * 16 + (lane & 15);
    const int kA = (lane >> 4) * 16;
    const int kB = (lane >> 4) * 16;
    const _Float16* Arow = hS + m * HDIM;

    if (t == 0) {
      // B operand from fp32 W_ih (converted in-register); A = f16(x).
      const float* Brow = W_ih + (size_t)(gate * HDIM + n0 + (lane & 15)) * HDIM;
      for (int k0 = 0; k0 < HDIM; k0 += 32) {
        v16h a = load_a16(Arow + k0 + kA);
        v16h b;
        const float4* bp = (const float4*)(Brow + k0 + kB);
#pragma unroll
        for (int j = 0; j < 4; ++j) {
          float4 f = bp[j];
          b[4 * j + 0] = (_Float16)f.x; b[4 * j + 1] = (_Float16)f.y;
          b[4 * j + 2] = (_Float16)f.z; b[4 * j + 3] = (_Float16)f.w;
        }
        acc = __builtin_amdgcn_wmma_f32_16x16x32_f16(false, a, false, b,
                                                     (short)0, acc, false, false);
      }
    } else {
      const _Float16* Brow = wS + (gate * 16 + (lane & 15)) * HDIM;
#pragma unroll 4
      for (int k0 = 0; k0 < HDIM; k0 += 32) {
        v16h a = load_a16(Arow + k0 + kA);
        v16h b = *(const v16h*)(Brow + k0 + kB);
        acc = __builtin_amdgcn_wmma_f32_16x16x32_f16(false, a, false, b,
                                                     (short)0, acc, false, false);
      }
    }

    // Scatter gate tile to LDS for cross-wave cell update.
    {
      int c = lane & 15;
      int rbase = mt * 16 + ((lane & 16) ? 8 : 0);
#pragma unroll
      for (int r = 0; r < 8; ++r)
        gS[(gate * 64 + rbase + r) * 16 + c] = acc[r];
    }
    __syncthreads();

    // Elementwise LSTM cell for this block's 64x16 slice.
    for (int p = tid; p < BATCH * 16; p += NTHR) {
      int bi = p >> 4, nc = p & 15;
      float gi = gS[(0 * 64 + bi) * 16 + nc] + bcomb[0 * HDIM + n0 + nc];
      float gf = gS[(1 * 64 + bi) * 16 + nc] + bcomb[1 * HDIM + n0 + nc];
      float gg = gS[(2 * 64 + bi) * 16 + nc] + bcomb[2 * HDIM + n0 + nc];
      float go = gS[(3 * 64 + bi) * 16 + nc] + bcomb[3 * HDIM + n0 + nc];
      float cn = sigmoidf_(gf) * cS[p] + sigmoidf_(gi) * tanhf(gg);
      float hn = sigmoidf_(go) * tanhf(cn);
      cS[p] = cn;
      _Float16 h16 = (_Float16)hn;
      hout[bi * HDIM + n0 + nc] = h16;
      seq16[((size_t)bi * TSTEPS + t) * HDIM + n0 + nc] = h16;
    }

    // Grid-wide step barrier (monotonic counter; 64 resident blocks).
    __syncthreads();
    if (tid == 0) {
      __threadfence();
      atomicAdd(sync, 1u);
      unsigned target = (unsigned)gridDim.x * (unsigned)(t + 1);
      while (__hip_atomic_load(sync, __ATOMIC_RELAXED, __HIP_MEMORY_SCOPE_AGENT) < target) {
        __builtin_amdgcn_s_sleep(1);
      }
      __threadfence();
    }
    __syncthreads();
  }
}

// ----- LayerNorm over H per (b,t) row ---------------------------------------

__global__ __launch_bounds__(256)
void ln_kernel(const _Float16* __restrict__ seq,
               const float* __restrict__ gamma, const float* __restrict__ beta,
               _Float16* __restrict__ seqn) {
  __shared__ float rs[256], rq[256];
  int row = blockIdx.x;
  const _Float16* p = seq + (size_t)row * HDIM;
  float s = 0.f, q = 0.f;
  for (int k = threadIdx.x; k < HDIM; k += 256) {
    float v = (float)p[k]; s += v; q += v * v;
  }
  rs[threadIdx.x] = s; rq[threadIdx.x] = q;
  __syncthreads();
  for (int off = 128; off > 0; off >>= 1) {
    if (threadIdx.x < off) {
      rs[threadIdx.x] += rs[threadIdx.x + off];
      rq[threadIdx.x] += rq[threadIdx.x + off];
    }
    __syncthreads();
  }
  float mu  = rs[0] * (1.f / HDIM);
  float var = rq[0] * (1.f / HDIM) - mu * mu;
  float inv = rsqrtf(var + 1e-5f);
  for (int k = threadIdx.x; k < HDIM; k += 256) {
    float v = (float)p[k];
    seqn[(size_t)row * HDIM + k] = (_Float16)((v - mu) * inv * gamma[k] + beta[k]);
  }
}

// ----- Output projection GEMM: [16384 x 1024] x [1024 x 512] ----------------

__global__ __launch_bounds__(256, 2)
void out_gemm(const _Float16* __restrict__ A,   // seqn [M][1024]
              const _Float16* __restrict__ Bm,  // Wout f16 [512][1024]
              const float* __restrict__ bias,   // [512]
              float* __restrict__ out) {        // [M][512]
  int wave = threadIdx.x >> 5, lane = threadIdx.x & 31;
  int Mb = blockIdx.x * 128 + wave * 16;
  int Nb = blockIdx.y * 16;
  const _Float16* Arow = A  + (size_t)(Mb + (lane & 15)) * HDIM;
  const _Float16* Brow = Bm + (size_t)(Nb + (lane & 15)) * HDIM;
  int kA = (lane >> 4) * 16, kB = (lane >> 4) * 16;
  v8f acc = {};
#pragma unroll 4
  for (int k0 = 0; k0 < HDIM; k0 += 32) {
    v16h a = load_a16(Arow + k0 + kA);
    v16h b = *(const v16h*)(Brow + k0 + kB);
    acc = __builtin_amdgcn_wmma_f32_16x16x32_f16(false, a, false, b,
                                                 (short)0, acc, false, false);
  }
  int col = Nb + (lane & 15);
  float bo = bias[col];
  int rbase = Mb + ((lane & 16) ? 8 : 0);
#pragma unroll
  for (int r = 0; r < 8; ++r)
    out[(size_t)(rbase + r) * DDIM + col] = acc[r] + bo;
}

// ----- launch ---------------------------------------------------------------

extern "C" void kernel_launch(void* const* d_in, const int* in_sizes, int n_in,
                              void* d_out, int out_size, void* d_ws, size_t ws_size,
                              hipStream_t stream) {
  const float* x     = (const float*)d_in[0];
  const float* W_ih  = (const float*)d_in[1];
  const float* W_hh  = (const float*)d_in[2];
  const float* b_ih  = (const float*)d_in[3];
  const float* b_hh  = (const float*)d_in[4];
  const float* gamma = (const float*)d_in[5];
  const float* beta  = (const float*)d_in[6];
  const float* W_out = (const float*)d_in[7];
  const float* b_out = (const float*)d_in[8];

  char* ws = (char*)d_ws;
  _Float16* Wc    = (_Float16*)(ws + OFF_WC);
  _Float16* seq16 = (_Float16*)(ws + OFF_SEQ);
  _Float16* seqn  = (_Float16*)(ws + OFF_SEQN);
  _Float16* Wo16  = (_Float16*)(ws + OFF_WO);
  _Float16* hG0   = (_Float16*)(ws + OFF_HG0);
  _Float16* hG1   = (_Float16*)(ws + OFF_HG1);
  unsigned* sync  = (unsigned*)(ws + OFF_SYNC);
  float*    bcomb = (float*)(ws + OFF_BCOMB);

  prep_wcomb<<<(GDIM * HDIM) / 256, 256, 0, stream>>>(W_ih, W_hh, Wc, GDIM * HDIM);
  prep_misc<<<(DDIM * HDIM) / 256, 256, 0, stream>>>(x, b_ih, b_hh, W_out,
                                                     hG0, bcomb, Wo16, sync);

  (void)hipFuncSetAttribute((const void*)lstm_persistent,
                            hipFuncAttributeMaxDynamicSharedMemorySize, SM_TOT);
  lstm_persistent<<<NBLK, NTHR, SM_TOT, stream>>>(W_ih, Wc, bcomb, hG0, hG1,
                                                  seq16, sync);

  ln_kernel<<<BATCH * TSTEPS, 256, 0, stream>>>(seq16, gamma, beta, seqn);

  dim3 g((BATCH * TSTEPS) / 128, DDIM / 16);
  out_gemm<<<g, 256, 0, stream>>>(seqn, Wo16, b_out, (float*)d_out);
}
